// KNRM_5626407158053
// MI455X (gfx1250) — matrix operand order repaired and checked by reading, not compiled, not run on hardware
//
#include <hip/hip_runtime.h>

#define LQ   64
#define LD   256
#define EMB  128
#define NK   21
#define HSTR 136   // padded f16 row stride (272B): breaks 16-way LDS bank conflicts

typedef _Float16 v16h __attribute__((ext_vector_type(16)));
typedef _Float16 v8h  __attribute__((ext_vector_type(8)));
typedef _Float16 v4h  __attribute__((ext_vector_type(4)));
typedef float    v8f  __attribute__((ext_vector_type(8)));

// LDS carve offsets (bytes)
#define OFF_Q   0
#define OFF_D   (LQ * HSTR * 2)                       // 17408
#define OFF_S   (OFF_D + LD * HSTR * 2)               // 87040
#define OFF_KM  (OFF_S + LQ * NK * 4)                 // 92416
#define SMEM_BYTES (OFF_KM + 128)                     // 92544

__device__ __forceinline__ v16h cat8(v8h a, v8h b) {
    return __builtin_shufflevector(a, b, 0,1,2,3,4,5,6,7,8,9,10,11,12,13,14,15);
}

__global__ __launch_bounds__(256) void knrm_fused(
    const int*   __restrict__ query, const int*   __restrict__ doc,
    const float* __restrict__ emb,
    const float* __restrict__ W1, const float* __restrict__ b1,
    const float* __restrict__ W2, const float* __restrict__ b2,
    const float* __restrict__ W3, const float* __restrict__ b3,
    float* __restrict__ out)
{
    extern __shared__ char smem[];
    _Float16* qh   = (_Float16*)(smem + OFF_Q);   // [64][136] f16, normalized query emb
    _Float16* dhm  = (_Float16*)(smem + OFF_D);   // [256][136] f16, normalized doc emb
    float*    ssum = (float*)   (smem + OFF_S);   // [64][21] doc-sum of kernel values
    float*    km   = (float*)   (smem + OFF_KM);  // [21]

    const int tid = threadIdx.x;
    const int b   = blockIdx.x;

    // ---- zero the (q, kernel) accumulator table ----
    for (int i = tid; i < LQ * NK; i += 256) ssum[i] = 0.f;

    // ---- gather + L2-normalize + f16-pack into LDS (two-pass, table is L2-resident) ----
    for (int tok = tid; tok < LQ + LD; tok += 256) {
        int idx; _Float16* dst;
        if (tok < LQ) { idx = query[b * LQ + tok];        dst = qh  + tok * HSTR; }
        else          { int t = tok - LQ;
                        idx = doc[b * LD + t];            dst = dhm + t  * HSTR; }
        const float4* src = (const float4*)(emb + (size_t)idx * EMB);
        float ss = 0.f;
        #pragma unroll
        for (int j = 0; j < EMB / 4; ++j) {
            float4 v = src[j];
            ss += v.x * v.x + v.y * v.y + v.z * v.z + v.w * v.w;
        }
        // x / max(||x||, 1e-12): padding row 0 is all-zero -> stays zero
        float inv = __builtin_amdgcn_rcpf(fmaxf(__builtin_amdgcn_sqrtf(ss), 1e-12f));
        v4h* d4 = (v4h*)dst;
        #pragma unroll
        for (int j = 0; j < EMB / 4; ++j) {
            float4 v = src[j];
            v4h h;
            h[0] = (_Float16)(v.x * inv); h[1] = (_Float16)(v.y * inv);
            h[2] = (_Float16)(v.z * inv); h[3] = (_Float16)(v.w * inv);
            d4[j] = h;                                   // ds_store_b64
        }
    }
    __syncthreads();

    // ---- transposed GEMM (mm^T tiles) + fused Gaussian kernel pooling ----
    const int lane = tid & 31;
    const int wave = tid >> 5;
    const int qt   = wave >> 1;          // query tile 0..3 (16 query rows each)
    const int dt0  = (wave & 1) * 8;     // this wave's 8 doc tiles (half of 16)

    // B fragment (32x16 f16, K x N): col = lane%16 (query token), lanes 0-15 K=0..15,
    // lanes 16-31 K=16..31, pairs packed per VGPR. Loaded once, reused for all doc tiles.
    const int ncol = lane & 15;
    const int kbB  = (lane & 16) ? 16 : 0;
    const _Float16* qrow = qh + (qt * 16 + ncol) * HSTR;
    v16h bfr[4];
    #pragma unroll
    for (int s = 0; s < 4; ++s) {
        const _Float16* p = qrow + 32 * s + kbB;
        bfr[s] = cat8(*(const v8h*)p, *(const v8h*)(p + 8));   // 2x ds_load_b128
    }

    float acc[NK];
    #pragma unroll
    for (int k = 0; k < NK; ++k) acc[k] = 0.f;

    // A fragment (16x32 f16, M x K): row = lane%16 (doc token); lanes 0-15 K=0..15,
    // lanes 16-31 K=16..31
    const int arow = lane & 15;
    const int kbA  = (lane & 16) ? 16 : 0;

    // A-fragment loader: 2x ds_load_b128 per K-step
    auto loadA = [&](int dt, v16h* f) {
        const _Float16* drow = dhm + (dt * 16 + arow) * HSTR;
        #pragma unroll
        for (int s = 0; s < 4; ++s) {
            const _Float16* p0 = drow + 32 * s + kbA;
            f[s] = cat8(*(const v8h*)p0, *(const v8h*)(p0 + 8));
        }
    };

    // GEMM tile + 21-kernel pooling epilogue (long VALU/exp tail hides LDS latency
    // of the next tile's prefetched A fragments)
    auto tile = [&](const v16h* af) {
        v8f c = {0.f, 0.f, 0.f, 0.f, 0.f, 0.f, 0.f, 0.f};
        #pragma unroll
        for (int s = 0; s < 4; ++s)
            c = __builtin_amdgcn_wmma_f32_16x16x32_f16(
                    false, af[s], false, bfr[s], (short)0, c, false, false);
        // c[v] = mm[doc = dt*16 + v + 8*(lane>=16)][query = qt*16 + lane%16]
        #pragma unroll
        for (int v = 0; v < 8; ++v) {
            float x = c[v];
            #pragma unroll
            for (int k = 0; k < NK - 1; ++k) {
                float mu = -0.95f + 0.1f * (float)k;            // sigma = 0.1
                float d0 = x - mu;
                acc[k] += __builtin_amdgcn_exp2f(-72.13474f * d0 * d0);  // -1/(2s^2)*log2e
            }
            float de = x - 1.0f;                                // exact kernel, sigma=0.001
            acc[NK - 1] += __builtin_amdgcn_exp2f(-721347.5f * de * de);
        }
    };

    // software-pipelined, double-buffered over 8 doc tiles (2 per iteration).
    // The tail prefetch (dt0+8 on the last iteration) reads past d_lds but stays
    // inside the allocated smem block and is never consumed.
    v16h a0[4], a1[4];
    loadA(dt0, a0);
    #pragma unroll 1
    for (int i = 0; i < 4; ++i) {
        const int dt = dt0 + 2 * i;
        loadA(dt + 1, a1);       // prefetch next tile before consuming a0
        tile(a0);                // wmma + pooling on tile dt
        loadA(dt + 2, a0);       // prefetch tile after next
        tile(a1);                // wmma + pooling on tile dt+1
    }

    // combine doc-row halves held by lane l and lane l^16
    #pragma unroll
    for (int k = 0; k < NK; ++k) acc[k] += __shfl_xor(acc[k], 16, 32);

    // lanes 0..15 each own one query row of this wave's tile -> LDS float atomics
    if (lane < 16) {
        float* srow = ssum + (qt * 16 + lane) * NK;
        #pragma unroll
        for (int k = 0; k < NK; ++k) atomicAdd(&srow[k], acc[k]);
    }
    __syncthreads();

    // ---- km[k] = relu( sum_q log1p(doc_sum) ) ----
    if (tid < NK) {
        float s = 0.f;
        for (int q = 0; q < LQ; ++q)
            s += 0.6931471805599453f *
                 __builtin_amdgcn_logf(1.0f + ssum[q * NK + tid]);  // v_log_f32
        km[tid] = fmaxf(s, 0.f);
    }
    __syncthreads();

    // ---- tiny MLP: 21 -> 10 (relu) -> 5 -> 1 ----
    if (tid == 0) {
        float x1[10];
        #pragma unroll
        for (int j = 0; j < 10; ++j) {
            float h = b1[j];
            for (int k = 0; k < NK; ++k) h += W1[j * NK + k] * km[k];
            x1[j] = fmaxf(h, 0.f);
        }
        float x2[5];
        #pragma unroll
        for (int i = 0; i < 5; ++i) {
            float h = b2[i];
            for (int j = 0; j < 10; ++j) h += W2[i * 10 + j] * x1[j];
            x2[i] = h;
        }
        float o = b3[0];
        #pragma unroll
        for (int i = 0; i < 5; ++i) o += W3[i] * x2[i];
        out[b] = o;
    }
}

extern "C" void kernel_launch(void* const* d_in, const int* in_sizes, int n_in,
                              void* d_out, int out_size, void* d_ws, size_t ws_size,
                              hipStream_t stream) {
    (void)n_in; (void)out_size; (void)d_ws; (void)ws_size;
    const int*   query = (const int*)  d_in[0];
    const int*   doc   = (const int*)  d_in[1];
    const float* emb   = (const float*)d_in[2];
    const float* W1    = (const float*)d_in[3];
    const float* b1    = (const float*)d_in[4];
    const float* W2    = (const float*)d_in[5];
    const float* b2    = (const float*)d_in[6];
    const float* W3    = (const float*)d_in[7];
    const float* b3    = (const float*)d_in[8];
    float*       out   = (float*)d_out;

    const int B = in_sizes[0] / LQ;   // 1024

    // dynamic LDS > 64KB: raise the per-kernel cap
    (void)hipFuncSetAttribute((const void*)knrm_fused,
                              hipFuncAttributeMaxDynamicSharedMemorySize,
                              (int)SMEM_BYTES);

    knrm_fused<<<dim3(B), dim3(256), SMEM_BYTES, stream>>>(
        query, doc, emb, W1, b1, W2, b2, W3, b3, out);
}
